// Encoder_39264591020550
// MI455X (gfx1250) — matrix operand order repaired and checked by reading, not compiled, not run on hardware
//
#include <hip/hip_runtime.h>

// ---------------------------------------------------------------------------
// LSTM encoder for MI455X (gfx1250, wave32, WMMA).
//   Prep   : one-shot f32->bf16 of embed/W_ih/W_hh + bias fold (b_ih+b_hh)
//   Phase 1: xg fragments = embed_bf[tok] @ W_ih^T + bias, stored in raw
//            WMMA C-layout blobs [t][bblk][jt][lane][8] (coalesced v8f)
//   Phase 2: 16 batch rows per workgroup run the full T=1024 recurrence
//            locally; h double-buffered in LDS (bf16); W_hh fragments pinned
//            in VGPRs (reg-residency forced); split accumulators halve the
//            WMMA chain depth; next-step xg tile prefetched each iteration.
// Matrix path: v_wmma_f32_16x16x32_bf16 (bf16 A/B, f32 accumulate).
// ---------------------------------------------------------------------------

typedef __attribute__((ext_vector_type(16))) __bf16 v16bf;
typedef __attribute__((ext_vector_type(8)))  __bf16 v8bf;
typedef __attribute__((ext_vector_type(4)))  __bf16 v4bf;
typedef __attribute__((ext_vector_type(8)))  float  v8f;

#define VSZ   50000
#define EDIM  128
#define HDIM  128
#define GDIM  512           // 4*H
#define BSZ   256
#define TSZ   1024
#define NBBLK (BSZ/16)      // 16 batch blocks
#define NJT   (GDIM/16)     // 32 gate-column tiles
#define TILEF (NJT*32*8)    // floats per (t,bblk) fragment blob = 8192

// --- prep kernels -----------------------------------------------------------
__global__ __launch_bounds__(256)
void cvt_f32_to_bf16(const float* __restrict__ src, __bf16* __restrict__ dst, int n4) {
    const int i = blockIdx.x * 256 + threadIdx.x;
    if (i < n4) {
        float4 v = ((const float4*)src)[i];
        v4bf r;
        r[0] = (__bf16)v.x; r[1] = (__bf16)v.y;
        r[2] = (__bf16)v.z; r[3] = (__bf16)v.w;
        ((v4bf*)dst)[i] = r;
    }
}

__global__ __launch_bounds__(256)
void bias_fold_kernel(const float* __restrict__ b_ih, const float* __restrict__ b_hh,
                      float* __restrict__ bias) {
    const int i = blockIdx.x * 256 + threadIdx.x;
    if (i < GDIM) bias[i] = b_ih[i] + b_hh[i];
}

// A-fragment from a bf16 row in LDS.  Contiguous K layout (matches B):
//   lanes 0-15 : K = k0..k0+15 (elems 0..15)
//   lanes 16-31: same +16 (via khalf)
__device__ inline v16bf buildA(const __bf16* hrow, int kc, int khalf) {
    const int k0 = kc*32 + khalf*16;
    v8bf a0 = *(const v8bf*)(hrow + k0);
    v8bf a1 = *(const v8bf*)(hrow + k0 + 8);
    v16bf r;
#pragma unroll
    for (int i = 0; i < 8; ++i) { r[i] = a0[i]; r[8+i] = a1[i]; }
    return r;
}

__device__ inline float sigmoid_(float x) { return 1.0f / (1.0f + __expf(-x)); }
__device__ inline float tanh_(float x)    { return 1.0f - 2.0f / (__expf(2.0f*x) + 1.0f); }

// --- Phase 1: gate precompute ----------------------------------------------
// grid = (B*T)/16 blocks of 256 threads (8 waves).  Block handles 16 (b,t)
// rows at fixed t; wave w handles N-tiles jt = 4w..4w+3.
__global__ __launch_bounds__(256)
void lstm_gates_kernel(const int* __restrict__ tokens,
                       const __bf16* __restrict__ embed_bf,
                       const __bf16* __restrict__ wih_bf,
                       const float* __restrict__ bias,
                       float* __restrict__ xg) {
    __shared__ __bf16 x_lds[16 * EDIM];

    const int tid   = threadIdx.x;
    const int lane  = tid & 31;
    const int wave  = tid >> 5;
    const int mtile = blockIdx.x;
    const int bblk  = mtile % NBBLK;
    const int t     = mtile / NBBLK;
    const int n     = lane & 15;
    const int khalf = lane >> 4;

    // stage 16 bf16 embedding rows (gather) into LDS: pure b128 copies
    {
        const int r  = tid >> 4;
        const int co = (tid & 15) * 8;
        const int b  = bblk*16 + r;
        const int tok = tokens[(long)b * TSZ + t];
        *(v8bf*)&x_lds[r*EDIM + co] = *(const v8bf*)(embed_bf + (long)tok*EDIM + co);
    }
    __syncthreads();

    v8f acc[4];
#pragma unroll
    for (int j = 0; j < 4; ++j) {
        const int g = (wave*4 + j)*16 + n;
        const float bv = bias[g];
#pragma unroll
        for (int r = 0; r < 8; ++r) acc[j][r] = bv;
    }

    // B fragments streamed from L2/L0 (occupancy hides latency here)
#pragma unroll
    for (int kc = 0; kc < 4; ++kc) {
        v16bf a = buildA(&x_lds[(lane & 15)*EDIM], kc, khalf);
#pragma unroll
        for (int j = 0; j < 4; ++j) {
            const int g = (wave*4 + j)*16 + n;
            v16bf bfr = *(const v16bf*)(wih_bf + (long)g*EDIM + kc*32 + khalf*16);
            acc[j] = __builtin_amdgcn_wmma_f32_16x16x32_bf16(
                false, a, false, bfr, (short)0, acc[j], false, false);
        }
    }

    // store raw C fragments: xg[t][bblk][jt][lane][8]  (32 B per lane, coalesced)
    float* blob = xg + ((long)t*NBBLK + bblk) * TILEF;
#pragma unroll
    for (int j = 0; j < 4; ++j) {
        const int jt = wave*4 + j;
        *(v8f*)(blob + ((long)jt*32 + lane)*8) = acc[j];
    }
}

// --- Phase 2: recurrence ----------------------------------------------------
// grid = B/16 = 16 blocks of 256 threads.  Wave w owns hidden slice
// [16w,16w+16) across all four gates (jt = gt*8 + w), so gate math and the
// c-update are wave-local in matching C layout.  h double-buffered in LDS ->
// one barrier per step.  Split accumulators: kc={0,1} into zero-init accB
// (starts right after the A ds_load), kc={2,3} into xg-init accA.
__global__ __launch_bounds__(256)
void lstm_scan_kernel(const float* __restrict__ xg,
                      const __bf16* __restrict__ whh_bf,
                      float* __restrict__ out) {
    __shared__ __bf16 h_lds[2 * 16 * HDIM];

    const int tid   = threadIdx.x;
    const int lane  = tid & 31;
    const int wave  = tid >> 5;
    const int bblk  = blockIdx.x;
    const int n     = lane & 15;
    const int khalf = lane >> 4;

    for (int i = tid; i < 16*HDIM; i += 256) h_lds[i] = (__bf16)0.0f;  // h(-1)=0

    // loop-invariant B fragments of W_hh^T: gate gt, slice col = gt*128+16w+n.
    // Pin them in VGPRs: the empty asm makes each fragment opaque so the
    // compiler cannot rematerialize it as a per-step global load on the
    // serial critical path (only 2 waves/SIMD here -> no occupancy cost).
    v16bf bfrag[4][4];
#pragma unroll
    for (int gt = 0; gt < 4; ++gt) {
        const int row = gt*HDIM + wave*16 + n;
#pragma unroll
        for (int kc = 0; kc < 4; ++kc) {
            bfrag[gt][kc] = *(const v16bf*)(whh_bf + (long)row*HDIM + kc*32 + khalf*16);
            asm volatile("" : "+v"(bfrag[gt][kc]));
        }
    }

    v8f c_state, h_reg;
#pragma unroll
    for (int r = 0; r < 8; ++r) { c_state[r] = 0.0f; h_reg[r] = 0.0f; }

    __syncthreads();

    for (int t = 0; t < TSZ; ++t) {
        const float* blob = xg + ((long)t*NBBLK + bblk) * TILEF;

        // prefetch next step's 32 KB blob (256 thr x 128 B)
        if (t + 1 < TSZ)
            __builtin_prefetch(blob + (long)NBBLK*TILEF + tid*32, 0, 0);

        const __bf16* hread  = h_lds + (t & 1) * (16*HDIM);
        __bf16*       hwrite = h_lds + ((t & 1) ^ 1) * (16*HDIM);

        // accA <- precomputed input-gate fragments (contiguous v8f per lane)
        v8f accA[4], accB[4];
#pragma unroll
        for (int gt = 0; gt < 4; ++gt) {
            const int jt = gt*8 + wave;
            accA[gt] = *(const v8f*)(blob + ((long)jt*32 + lane)*8);
#pragma unroll
            for (int r = 0; r < 8; ++r) accB[gt][r] = 0.0f;
        }

        // g += h @ W_hh^T ; two independent chains of depth 2 per gate
#pragma unroll
        for (int kc = 0; kc < 4; ++kc) {
            v16bf a = buildA(&hread[(lane & 15)*HDIM], kc, khalf);
#pragma unroll
            for (int gt = 0; gt < 4; ++gt) {
                if (kc < 2)
                    accB[gt] = __builtin_amdgcn_wmma_f32_16x16x32_bf16(
                        false, a, false, bfrag[gt][kc], (short)0, accB[gt], false, false);
                else
                    accA[gt] = __builtin_amdgcn_wmma_f32_16x16x32_bf16(
                        false, a, false, bfrag[gt][kc], (short)0, accA[gt], false, false);
            }
        }

#pragma unroll
        for (int r = 0; r < 8; ++r) {
            const float ig = sigmoid_(accA[0][r] + accB[0][r]);
            const float fg = sigmoid_(accA[1][r] + accB[1][r]);
            const float gg = tanh_   (accA[2][r] + accB[2][r]);
            const float og = sigmoid_(accA[3][r] + accB[3][r]);
            const float cn = fg*c_state[r] + ig*gg;
            c_state[r] = cn;
            const float hn = og * tanh_(cn);
            h_reg[r] = hn;
            const int m = r + 8*khalf;
            hwrite[m*HDIM + wave*16 + n] = (__bf16)hn;
        }

        __syncthreads();   // h(t) visible before step t+1 reads it
    }

    // final (h, c): out[0 .. B*H) = h, out[B*H .. 2*B*H) = c
#pragma unroll
    for (int r = 0; r < 8; ++r) {
        const int m = r + 8*khalf;
        const int b = bblk*16 + m;
        const int j = wave*16 + n;
        out[(long)b*HDIM + j]                  = h_reg[r];
        out[(long)BSZ*HDIM + (long)b*HDIM + j] = c_state[r];
    }
}

// --- launcher ---------------------------------------------------------------
extern "C" void kernel_launch(void* const* d_in, const int* in_sizes, int n_in,
                              void* d_out, int out_size, void* d_ws, size_t ws_size,
                              hipStream_t stream) {
    const int*   tokens = (const int*)  d_in[0];
    const float* embed  = (const float*)d_in[1];
    const float* W_ih   = (const float*)d_in[2];
    const float* W_hh   = (const float*)d_in[3];
    const float* b_ih   = (const float*)d_in[4];
    const float* b_hh   = (const float*)d_in[5];
    float* out = (float*)d_out;

    // workspace: xg f32 blobs | embed bf16 | W_ih bf16 | W_hh bf16 | bias f32
    char* ws = (char*)d_ws;
    const size_t XG_BYTES  = (size_t)TSZ * NBBLK * TILEF * sizeof(float);  // 512 MB
    const size_t EMB_BYTES = (size_t)VSZ * EDIM * sizeof(__bf16);          // 12.8 MB
    const size_t W_BYTES   = (size_t)GDIM * EDIM * sizeof(__bf16);         // 128 KB
    float*  xg       = (float*)ws;
    __bf16* embed_bf = (__bf16*)(ws + XG_BYTES);
    __bf16* wih_bf   = (__bf16*)(ws + XG_BYTES + EMB_BYTES);
    __bf16* whh_bf   = (__bf16*)(ws + XG_BYTES + EMB_BYTES + W_BYTES);
    float*  bias     = (float*) (ws + XG_BYTES + EMB_BYTES + 2*W_BYTES);

    // one-shot prep (stream-ordered before the hot kernels)
    {
        const int n4e = (VSZ * EDIM) / 4;
        cvt_f32_to_bf16<<<(n4e + 255)/256, 256, 0, stream>>>(embed, embed_bf, n4e);
        const int n4w = (GDIM * EDIM) / 4;
        cvt_f32_to_bf16<<<(n4w + 255)/256, 256, 0, stream>>>(W_ih, wih_bf, n4w);
        cvt_f32_to_bf16<<<(n4w + 255)/256, 256, 0, stream>>>(W_hh, whh_bf, n4w);
        bias_fold_kernel<<<(GDIM + 255)/256, 256, 0, stream>>>(b_ih, b_hh, bias);
    }

    lstm_gates_kernel<<<(BSZ*TSZ)/16, 256, 0, stream>>>(tokens, embed_bf, wih_bf, bias, xg);
    lstm_scan_kernel<<<BSZ/16, 256, 0, stream>>>(xg, whh_bf, out);
}
